// MILSTM_87978110091460
// MI455X (gfx1250) — compile-verified, matrix operand order checked by reading, not run on hardware
//
#include <hip/hip_runtime.h>
#include <math.h>

// ---------------------------------------------------------------------------
// MI-LSTM for MI455X (gfx1250, wave32, WMMA bf16, async global->LDS).
//   Phase 1: xp[t,b,:] = x[b,t,:] @ Wk + bk      (one big WMMA GEMM, bf16)
//   Phase 2: 256 fused step kernels: hp = h @ Wr (WMMA), MI gating + cell
//   Phase 3: out = h_T @ Wc + bc                 (WMMA GEMM)
// Each wave carries TWO accumulators (two 16-row M subtiles sharing one
// B fragment): 2-way WMMA ILP and half the L2 weight traffic.
// ---------------------------------------------------------------------------

#define UNITS  1024
#define DIN    1024
#define NCLASS 1024
#define BATCH  128
#define TSTEPS 256   // power of two: row remap is shift/mask
#define ZCOLS  4096  // 4*UNITS
#define BT     (BATCH * TSTEPS)

typedef __attribute__((ext_vector_type(16))) __bf16 v16bf;
typedef __attribute__((ext_vector_type(8)))  float  v8f;
typedef int v4i __attribute__((vector_size(4 * sizeof(int))));

union FragBF {           // 16 bf16 = 32 bytes = two uint4 chunks
    v16bf v;
    uint4 u[2];
};

#if defined(__HIP_DEVICE_COMPILE__) && __has_builtin(__builtin_amdgcn_global_load_async_to_lds_b128)
#define ASYNC_LDS 1
#else
#define ASYNC_LDS 0
#endif

// Copy 16 bytes global -> LDS. Async path bypasses VGPRs (ASYNCcnt).
__device__ __forceinline__ void stage16(const __bf16* g, __bf16* l) {
#if ASYNC_LDS
    __builtin_amdgcn_global_load_async_to_lds_b128(
        (__attribute__((address_space(1))) v4i*)(uintptr_t)g,
        (__attribute__((address_space(3))) v4i*)(uintptr_t)l, 0, 0);
#else
    *reinterpret_cast<uint4*>(l) = *reinterpret_cast<const uint4*>(g);
#endif
}

__device__ __forceinline__ void stage_wait_barrier() {
#if ASYNC_LDS
    asm volatile("s_wait_asynccnt 0x0" ::: "memory");
#endif
    __syncthreads();
}

__device__ __forceinline__ v8f wmma_bf16(const FragBF& a, const FragBF& b, v8f c) {
    return __builtin_amdgcn_wmma_f32_16x16x32_bf16(false, a.v, false, b.v,
                                                   (short)0, c, false, false);
}

// ------------------------------ helpers ------------------------------------

__global__ void to_bf16_kernel(const float* __restrict__ src,
                               __bf16* __restrict__ dst, int n) {
    int i = blockIdx.x * blockDim.x + threadIdx.x;
    if (i < n) dst[i] = (__bf16)src[i];
}

// Wt[n][k] = (bf16) W[k][n]   (W is [K,N] row-major)
__global__ void transpose_bf16_kernel(const float* __restrict__ W,
                                      __bf16* __restrict__ Wt, int K, int N) {
    int i = blockIdx.x * blockDim.x + threadIdx.x;
    if (i >= N * K) return;
    int n = i / K, k = i % K;
    Wt[i] = (__bf16)W[(size_t)k * N + n];
}

__global__ void init_hc_kernel(const float* __restrict__ h0,
                               const float* __restrict__ c0,
                               __bf16* __restrict__ hb,
                               float* __restrict__ cb, int n) {
    int i = blockIdx.x * blockDim.x + threadIdx.x;
    if (i < n) { hb[i] = (__bf16)h0[i]; cb[i] = c0[i]; }
}

// ------------------------- generic WMMA GEMM -------------------------------
// C[M,N] = A[M,K] (bf16 row-major) * Bt[N,K] (bf16, pre-transposed) + bias[N]
// Block = 128 threads (4 waves). Block tile = 32 rows x 64 cols; each wave
// owns two stacked 16x16 tiles (one B frag -> 2 WMMAs). A staged in LDS via
// async global->LDS, K chunked by 64. If remap!=0: input row r = b*TSTEPS+t
// is written to output row t*BATCH+b (shift/mask, TSTEPS is a power of two).
__global__ __launch_bounds__(128)
void gemm_bf16_kernel(const __bf16* __restrict__ A,
                      const __bf16* __restrict__ Bt,
                      const float*  __restrict__ bias,
                      float*  __restrict__ Cf,    // optional f32 out
                      __bf16* __restrict__ Cbf,   // optional bf16 out
                      int M, int N, int K, int remap)
{
    __shared__ __bf16 As[32][72];                // +8 pad: bank decorrelation

    const int tid  = threadIdx.x;
    const int wave = tid >> 5;
    const int lane = tid & 31;
    const int mbase = blockIdx.y * 32;
    const int nbase = blockIdx.x * 64 + wave * 16;

    const int row = tid >> 3;                    // 0..15  (A stage, 2 rows each)
    const int seg = tid & 7;                     // 0..7   (16B segments)

    const int n     = nbase + (lane & 15);
    const int kbase = (lane >> 4) * 8;           // ISA A/B lane-half K split
    const __bf16* Brow = Bt + (size_t)n * K;

    v8f acc0 = {}, acc1 = {};
    for (int k0 = 0; k0 < K; k0 += 64) {
        stage16(A + (size_t)(mbase + row) * K + k0 + seg * 8,      &As[row][seg * 8]);
        stage16(A + (size_t)(mbase + row + 16) * K + k0 + seg * 8, &As[row + 16][seg * 8]);
        __builtin_prefetch(Brow + k0 + 64, 0, 1);   // global_prefetch next B chunk
        stage_wait_barrier();
#pragma unroll
        for (int kk = 0; kk < 64; kk += 32) {
            FragBF a0, a1, b;
            b.u[0]  = *reinterpret_cast<const uint4*>(Brow + k0 + kk + kbase);
            b.u[1]  = *reinterpret_cast<const uint4*>(Brow + k0 + kk + kbase + 16);
            a0.u[0] = *reinterpret_cast<const uint4*>(&As[lane & 15][kk + kbase]);
            a0.u[1] = *reinterpret_cast<const uint4*>(&As[lane & 15][kk + kbase + 16]);
            a1.u[0] = *reinterpret_cast<const uint4*>(&As[16 + (lane & 15)][kk + kbase]);
            a1.u[1] = *reinterpret_cast<const uint4*>(&As[16 + (lane & 15)][kk + kbase + 16]);
            acc0 = wmma_bf16(a0, b, acc0);
            acc1 = wmma_bf16(a1, b, acc1);
        }
        __syncthreads();
    }

    const float bv = bias ? bias[n] : 0.0f;
#pragma unroll
    for (int v = 0; v < 8; ++v) {
        int ml = v + (lane >> 4) * 8;            // ISA C/D layout
#pragma unroll
        for (int s = 0; s < 2; ++s) {
            int   m   = mbase + ml + s * 16;
            float val = (s ? acc1[v] : acc0[v]) + bv;
            int orow = m;
            if (remap) {                          // constant shift/mask remap
                int bi = m >> 8;                  // m / TSTEPS
                int t  = m & (TSTEPS - 1);        // m % TSTEPS
                orow = t * BATCH + bi;
            }
            size_t o = (size_t)orow * N + n;
            if (Cf)  Cf[o]  = val;
            if (Cbf) Cbf[o] = (__bf16)val;
        }
    }
}

// ------------------------- fused MI-LSTM step ------------------------------
// Block = 128 threads (4 waves); wave g computes gate g's two stacked 16x16
// hp tiles (32 batch rows x 16 units), then the cell update fuses via LDS.
__global__ __launch_bounds__(128)
void milstm_step_kernel(const __bf16* __restrict__ h_in,   // [B,U] bf16
                        const float*  __restrict__ c_in,   // [B,U]
                        const __bf16* __restrict__ WrT,    // [4U,U] bf16
                        const float*  __restrict__ br,     // [4U]
                        const __bf16* __restrict__ xp_t,   // [B,4U] bf16, step t
                        const float*  __restrict__ alpha,
                        const float*  __restrict__ beta1,
                        const float*  __restrict__ beta2,
                        __bf16* __restrict__ h_out,        // [B,U] bf16
                        float*  __restrict__ hf_out,       // [B,U] f32
                        float*  __restrict__ c_out)        // [B,U]
{
    __shared__ __bf16 Hs[32][72];
    __shared__ float  zs[4][32][16];

    const int tid  = threadIdx.x;
    const int gate = tid >> 5;
    const int lane = tid & 31;
    const int bbase = blockIdx.y * 32;   // batch rows
    const int ubase = blockIdx.x * 16;   // unit cols

    const int row = tid >> 3;
    const int seg = tid & 7;

    const int nn    = ubase + (lane & 15);
    const int zcol  = gate * UNITS + nn;
    const int kbase = (lane >> 4) * 8;
    const __bf16* Brow = WrT + (size_t)zcol * UNITS;

    v8f acc0 = {}, acc1 = {};
    for (int k0 = 0; k0 < UNITS; k0 += 64) {
        stage16(h_in + (size_t)(bbase + row) * UNITS + k0 + seg * 8,      &Hs[row][seg * 8]);
        stage16(h_in + (size_t)(bbase + row + 16) * UNITS + k0 + seg * 8, &Hs[row + 16][seg * 8]);
        __builtin_prefetch(Brow + k0 + 64, 0, 1);
        stage_wait_barrier();
#pragma unroll
        for (int kk = 0; kk < 64; kk += 32) {
            FragBF a0, a1, b;
            b.u[0]  = *reinterpret_cast<const uint4*>(Brow + k0 + kk + kbase);
            b.u[1]  = *reinterpret_cast<const uint4*>(Brow + k0 + kk + kbase + 16);
            a0.u[0] = *reinterpret_cast<const uint4*>(&Hs[lane & 15][kk + kbase]);
            a0.u[1] = *reinterpret_cast<const uint4*>(&Hs[lane & 15][kk + kbase + 16]);
            a1.u[0] = *reinterpret_cast<const uint4*>(&Hs[16 + (lane & 15)][kk + kbase]);
            a1.u[1] = *reinterpret_cast<const uint4*>(&Hs[16 + (lane & 15)][kk + kbase + 16]);
            acc0 = wmma_bf16(a0, b, acc0);
            acc1 = wmma_bf16(a1, b, acc1);
        }
        __syncthreads();
    }

    // z = alpha*xp*hp + beta1*xp + beta2*hp  (per element of this gate tile)
    const float brv = br[zcol];
    const float al  = alpha[zcol];
    const float b1  = beta1[zcol];
    const float b2  = beta2[zcol];
#pragma unroll
    for (int v = 0; v < 8; ++v) {
        int ml = v + (lane >> 4) * 8;
#pragma unroll
        for (int s = 0; s < 2; ++s) {
            int   m  = ml + s * 16;
            float hp = (s ? acc1[v] : acc0[v]) + brv;
            float xv = (float)xp_t[(size_t)(bbase + m) * ZCOLS + zcol];
            zs[gate][m][lane & 15] = al * xv * hp + b1 * xv + b2 * hp;
        }
    }
    __syncthreads();

    // cell update: 512 elements / 128 threads
#pragma unroll
    for (int rep = 0; rep < 4; ++rep) {
        int e = tid + rep * 128;
        int m = e >> 4, nu = e & 15;
        float zi = zs[0][m][nu], zf = zs[1][m][nu];
        float zg = zs[2][m][nu], zo = zs[3][m][nu];
        size_t idx = (size_t)(bbase + m) * UNITS + (ubase + nu);
        float cold = c_in[idx];
        float si = 1.0f / (1.0f + __expf(-zi));
        float sf = 1.0f / (1.0f + __expf(-zf));
        float so = 1.0f / (1.0f + __expf(-zo));
        float cn = tanhf(zg) * si + cold * sf;
        float hn = tanhf(cn) * so;
        c_out[idx]  = cn;
        hf_out[idx] = hn;
        h_out[idx]  = (__bf16)hn;
    }
}

// ------------------------------ launcher -----------------------------------

extern "C" void kernel_launch(void* const* d_in, const int* in_sizes, int n_in,
                              void* d_out, int out_size, void* d_ws, size_t ws_size,
                              hipStream_t stream) {
    const float* x     = (const float*)d_in[0];
    const float* h0    = (const float*)d_in[1];
    const float* c0    = (const float*)d_in[2];
    const float* Wk    = (const float*)d_in[3];
    const float* bk    = (const float*)d_in[4];
    const float* Wr    = (const float*)d_in[5];
    const float* br    = (const float*)d_in[6];
    const float* alpha = (const float*)d_in[7];
    const float* beta1 = (const float*)d_in[8];
    const float* beta2 = (const float*)d_in[9];
    const float* Wc    = (const float*)d_in[10];
    const float* bc    = (const float*)d_in[11];
    float* out = (float*)d_out;

    // workspace carve (256B aligned)
    char* p = (char*)d_ws;
    auto carve = [&](size_t bytes) -> void* {
        void* r = (void*)p; p += (bytes + 255) & ~(size_t)255; return r;
    };
    __bf16* xb  = (__bf16*)carve((size_t)BT * DIN * 2);
    __bf16* WkT = (__bf16*)carve((size_t)ZCOLS * DIN * 2);
    __bf16* WrT = (__bf16*)carve((size_t)ZCOLS * UNITS * 2);
    __bf16* WcT = (__bf16*)carve((size_t)NCLASS * UNITS * 2);
    __bf16* xp  = (__bf16*)carve((size_t)TSTEPS * BATCH * ZCOLS * 2);  // [T,B,4U]
    __bf16* hb[2]; float* cb[2];
    hb[0] = (__bf16*)carve((size_t)BATCH * UNITS * 2);
    hb[1] = (__bf16*)carve((size_t)BATCH * UNITS * 2);
    cb[0] = (float*)carve((size_t)BATCH * UNITS * 4);
    cb[1] = (float*)carve((size_t)BATCH * UNITS * 4);
    float* hf = (float*)carve((size_t)BATCH * UNITS * 4);

    // 1) precision demotion + weight transposes
    to_bf16_kernel<<<(BT * DIN + 255) / 256, 256, 0, stream>>>(x, xb, BT * DIN);
    transpose_bf16_kernel<<<(ZCOLS * DIN + 255) / 256, 256, 0, stream>>>(Wk, WkT, DIN, ZCOLS);
    transpose_bf16_kernel<<<(ZCOLS * UNITS + 255) / 256, 256, 0, stream>>>(Wr, WrT, UNITS, ZCOLS);
    transpose_bf16_kernel<<<(NCLASS * UNITS + 255) / 256, 256, 0, stream>>>(Wc, WcT, UNITS, NCLASS);
    init_hc_kernel<<<(BATCH * UNITS + 255) / 256, 256, 0, stream>>>(
        h0, c0, hb[0], cb[0], BATCH * UNITS);

    // 2) xp = x @ Wk + bk  for all timesteps, reordered to [T,B,4U]
    gemm_bf16_kernel<<<dim3(ZCOLS / 64, BT / 32), 128, 0, stream>>>(
        xb, WkT, bk, nullptr, xp, BT, ZCOLS, DIN, /*remap=*/1);

    // 3) sequential recurrence, one fused kernel per step
    for (int t = 0; t < TSTEPS; ++t) {
        milstm_step_kernel<<<dim3(UNITS / 16, BATCH / 32), 128, 0, stream>>>(
            hb[t & 1], cb[t & 1], WrT, br,
            xp + (size_t)t * BATCH * ZCOLS,
            alpha, beta1, beta2,
            hb[(t + 1) & 1], hf, cb[(t + 1) & 1]);
    }
    const int fin = TSTEPS & 1;   // == 0

    // 4) classifier: out = h_T @ Wc + bc
    gemm_bf16_kernel<<<dim3(NCLASS / 64, BATCH / 32), 128, 0, stream>>>(
        hb[fin], WcT, bc, out, nullptr, BATCH, NCLASS, UNITS, /*remap=*/0);

    // 5) tuple outputs (out, h, c)
    (void)hipMemcpyAsync(out + (size_t)BATCH * NCLASS, hf,
                         (size_t)BATCH * UNITS * 4, hipMemcpyDeviceToDevice, stream);
    (void)hipMemcpyAsync(out + (size_t)BATCH * NCLASS + (size_t)BATCH * UNITS, cb[fin],
                         (size_t)BATCH * UNITS * 4, hipMemcpyDeviceToDevice, stream);
    (void)in_sizes; (void)n_in; (void)out_size; (void)ws_size;
}
